// BCMF_3496103379251
// MI455X (gfx1250) — matrix operand order repaired and checked by reading, not compiled, not run on hardware
//
#include <hip/hip_runtime.h>

// ---------------------------------------------------------------------------
// Types / helpers for CDNA5 (gfx1250) wave32 WMMA
// ---------------------------------------------------------------------------
typedef __attribute__((ext_vector_type(16))) __bf16 v16bf;
typedef __attribute__((ext_vector_type(8)))  float  v8f;

union Frag16 { v16bf v; uint4 q[2]; };

// Load a 16-element bf16 fragment as two 16-byte chunks (alias-safe memcpy,
// lowers to b128 loads given the 16B alignment we guarantee by construction).
__device__ __forceinline__ v16bf load_frag(const __bf16* p0, const __bf16* p1) {
  Frag16 f;
  const void* a0 = __builtin_assume_aligned(p0, 16);
  const void* a1 = __builtin_assume_aligned(p1, 16);
  __builtin_memcpy(&f.q[0], a0, 16);
  __builtin_memcpy(&f.q[1], a1, 16);
  return f.v;
}

__device__ __forceinline__ v8f wmma_bf16(v16bf a, v16bf b, v8f c) {
  // D = A(16x32 bf16) * B(32x16 bf16) + C(16x16 f32)
  return __builtin_amdgcn_wmma_f32_16x16x32_bf16(
      /*neg_a=*/false, a, /*neg_b=*/false, b,
      /*c_mod=*/(short)0, c, /*reuse_a=*/false, /*reuse_b=*/false);
}

// Reductions across a 16-lane half of the wave (xor<16 stays within half).
__device__ __forceinline__ float half16_max(float v) {
#pragma unroll
  for (int off = 1; off < 16; off <<= 1) v = fmaxf(v, __shfl_xor(v, off, 32));
  return v;
}
__device__ __forceinline__ float half16_sum(float v) {
#pragma unroll
  for (int off = 1; off < 16; off <<= 1) v += __shfl_xor(v, off, 32);
  return v;
}

// ---------------------------------------------------------------------------
// Kernel 1: projection GEMM  Y = (X(f32)[S,256] @ W(f32)[256,256]) * oscale
// MODE 0: write per-head row-major   out[((b*8+h)*SROWS + s)*32 + hd]
// MODE 1: write per-head transposed  out[((b*8+h)*32 + hd)*SROWS + s]
// Block: 128x64 output tile, 8 waves, each wave 2x2 of 16x16 WMMA tiles.
// ---------------------------------------------------------------------------
template <int MODE, int SROWS>
__global__ __launch_bounds__(256) void proj_gemm(const float* __restrict__ X,
                                                 const float* __restrict__ W,
                                                 __bf16* __restrict__ out,
                                                 float oscale) {
  __shared__ __align__(16) __bf16 As[128][64];  // A slab  (rows x k)
  __shared__ __align__(16) __bf16 Bt[64][64];   // B^T slab (cols x k)
  const int tid  = threadIdx.x;
  const int row0 = blockIdx.x * 128;
  const int col0 = blockIdx.y * 64;
  const int w    = tid >> 5, lane = tid & 31;
  const int ml   = lane & 15, hi = lane >> 4;
  const int wr   = (w & 3) * 32, wc = (w >> 2) * 32;

  v8f acc00 = {0.f,0.f,0.f,0.f,0.f,0.f,0.f,0.f};
  v8f acc01 = acc00, acc10 = acc00, acc11 = acc00;

  for (int ko = 0; ko < 256; ko += 64) {
    __syncthreads();
    // A slab: 128 rows x 64 k, convert f32 -> bf16
    for (int i = tid; i < 2048; i += 256) {
      int r = i >> 4, c4 = (i & 15) << 2;
      float4 v = *(const float4*)(X + (size_t)(row0 + r) * 256 + ko + c4);
      As[r][c4 + 0] = (__bf16)v.x; As[r][c4 + 1] = (__bf16)v.y;
      As[r][c4 + 2] = (__bf16)v.z; As[r][c4 + 3] = (__bf16)v.w;
    }
    // B slab: transpose+convert W[k][n] -> Bt[n][k]
    for (int i = tid; i < 1024; i += 256) {
      int k = i >> 4, j4 = (i & 15) << 2;
      float4 v = *(const float4*)(W + (size_t)(ko + k) * 256 + col0 + j4);
      Bt[j4 + 0][k] = (__bf16)v.x; Bt[j4 + 1][k] = (__bf16)v.y;
      Bt[j4 + 2][k] = (__bf16)v.z; Bt[j4 + 3][k] = (__bf16)v.w;
    }
    __syncthreads();
#pragma unroll
    for (int kk = 0; kk < 64; kk += 32) {
      v16bf a0 = load_frag(&As[wr + ml][kk + hi * 8],      &As[wr + ml][kk + 16 + hi * 8]);
      v16bf a1 = load_frag(&As[wr + 16 + ml][kk + hi * 8], &As[wr + 16 + ml][kk + 16 + hi * 8]);
      v16bf b0 = load_frag(&Bt[wc + ml][kk + hi * 8],      &Bt[wc + ml][kk + 16 + hi * 8]);
      v16bf b1 = load_frag(&Bt[wc + 16 + ml][kk + hi * 8], &Bt[wc + 16 + ml][kk + 16 + hi * 8]);
      acc00 = wmma_bf16(a0, b0, acc00);
      acc01 = wmma_bf16(a0, b1, acc01);
      acc10 = wmma_bf16(a1, b0, acc10);
      acc11 = wmma_bf16(a1, b1, acc11);
    }
  }

  constexpr int LG = (SROWS == 512) ? 9 : 12;
  v8f accs[2][2] = {{acc00, acc01}, {acc10, acc11}};
#pragma unroll
  for (int i = 0; i < 2; i++)
#pragma unroll
    for (int j = 0; j < 2; j++)
#pragma unroll
      for (int r = 0; r < 8; r++) {
        int s = row0 + wr + i * 16 + r + 8 * hi;  // C layout: row = r (+8 for hi half)
        int d = col0 + wc + j * 16 + ml;          // col = lane%16
        int b = s >> LG, sr = s & (SROWS - 1);
        int h = d >> 5, hd = d & 31;
        size_t idx = (MODE == 0)
                         ? (((size_t)(b * 8 + h) * SROWS + sr) * 32 + hd)
                         : (((size_t)(b * 8 + h) * 32 + hd) * SROWS + sr);
        out[idx] = (__bf16)(accs[i][j][r] * oscale);
      }
}

// ---------------------------------------------------------------------------
// Kernel 2/3: flash attention (one direction), 64-key chunks.
// Q  : [BH, Nq, 32] bf16 (row-major per head, pre-scaled by 1/sqrt(hd))
// K  : [BH, Nk, 32] bf16 (row-major per head)  -> B fragments contiguous
// Vt : [BH, 32, Nk] bf16 (transposed per head) -> B fragments contiguous
// ctx: [B, Nq, 256] bf16 merged-head output
// MASKED: key-mask (amr_pad_mask), masked score -> -3e38 (== finfo.min-ish)
// Each wave owns 16 query rows; online softmax amortized over 64 keys.
// ---------------------------------------------------------------------------
template <bool MASKED>
__global__ __launch_bounds__(256) void attn_kernel(
    const __bf16* __restrict__ Q, const __bf16* __restrict__ K,
    const __bf16* __restrict__ Vt, __bf16* __restrict__ ctx,
    const int* __restrict__ mask, int Nq, int Nk) {
  __shared__ __align__(16) __bf16 Pbuf[8][16 * 64];  // per-wave P scratch
  const int tid = threadIdx.x;
  const int w = tid >> 5, lane = tid & 31;
  const int ml = lane & 15, hi = lane >> 4;
  const int bh = blockIdx.y, b = bh >> 3, h = bh & 7;
  const int n0 = blockIdx.x * 128 + w * 16;

  const __bf16* Qh = Q + (size_t)bh * Nq * 32;
  const __bf16* Kh = K + (size_t)bh * Nk * 32;
  const __bf16* Vh = Vt + (size_t)bh * 32 * Nk;

  // Q as A-fragment: lane holds row n0+ml, k-halves selected by hi.
  const __bf16* qrow = Qh + (size_t)(n0 + ml) * 32;
  v16bf qa = load_frag(qrow + hi * 16, qrow + hi * 16 + 8);

  v8f o0 = {0.f,0.f,0.f,0.f,0.f,0.f,0.f,0.f};  // out cols hd 0..15
  v8f o1 = o0;                                  // out cols hd 16..31
  const v8f zero = o0;
  float m[8], l[8];
#pragma unroll
  for (int r = 0; r < 8; r++) { m[r] = -3.0e38f; l[r] = 0.f; }

  __bf16* P = &Pbuf[w][0];

  for (int m0 = 0; m0 < Nk; m0 += 64) {
    // Scores: four 16-key tiles (HD=32 => one WMMA each, K already in B-layout)
    v8f s[4];
#pragma unroll
    for (int t = 0; t < 4; t++) {
      const __bf16* kr = Kh + (size_t)(m0 + t * 16 + ml) * 32 + hi * 16;
      v16bf kb = load_frag(kr, kr + 8);
      s[t] = wmma_bf16(qa, kb, zero);
    }

    bool msk[4] = {false, false, false, false};
    if (MASKED) {
#pragma unroll
      for (int t = 0; t < 4; t++)
        msk[t] = mask[b * Nk + m0 + t * 16 + ml] != 0;
    }

#pragma unroll
    for (int r = 0; r < 8; r++) {
      float a[4];
#pragma unroll
      for (int t = 0; t < 4; t++) {
        a[t] = s[t][r];
        if (MASKED && msk[t]) a[t] = -3.0e38f;
      }
      float cm = half16_max(fmaxf(fmaxf(a[0], a[1]), fmaxf(a[2], a[3])));
      float mn = fmaxf(m[r], cm);
      float alpha = __expf(m[r] - mn);
      m[r] = mn;
      float p[4], ps = 0.f;
#pragma unroll
      for (int t = 0; t < 4; t++) { p[t] = __expf(a[t] - mn); ps += p[t]; }
      l[r] = l[r] * alpha + half16_sum(ps);
      o0[r] *= alpha;
      o1[r] *= alpha;
      int row = r + 8 * hi;  // C-layout row owned by this lane
#pragma unroll
      for (int t = 0; t < 4; t++) P[row * 64 + t * 16 + ml] = (__bf16)p[t];
    }

    // Re-read P as A-fragments (LDS is in-order per wave; region is private).
    v16bf pa0 = load_frag(P + ml * 64 + hi * 16,      P + ml * 64 + hi * 16 + 8);
    v16bf pa1 = load_frag(P + ml * 64 + 32 + hi * 16, P + ml * 64 + 32 + hi * 16 + 8);

    // V^T rows as B-fragments: col = hd, k = key chunk
    const __bf16* vr0 = Vh + (size_t)ml * Nk + m0 + hi * 16;
    const __bf16* vr1 = Vh + (size_t)(16 + ml) * Nk + m0 + hi * 16;
    v16bf vb00 = load_frag(vr0, vr0 + 8);        // keys m0..m0+31, hd 0..15
    v16bf vb01 = load_frag(vr1, vr1 + 8);        // keys m0..m0+31, hd 16..31
    v16bf vb10 = load_frag(vr0 + 32, vr0 + 40);  // keys m0+32..m0+63, hd 0..15
    v16bf vb11 = load_frag(vr1 + 32, vr1 + 40);  // keys m0+32..m0+63, hd 16..31
    o0 = wmma_bf16(pa0, vb00, o0);
    o1 = wmma_bf16(pa0, vb01, o1);
    o0 = wmma_bf16(pa1, vb10, o0);
    o1 = wmma_bf16(pa1, vb11, o1);
  }

#pragma unroll
  for (int r = 0; r < 8; r++) {
    float inv = 1.0f / l[r];
    int n = n0 + r + 8 * hi;
    size_t base = ((size_t)b * Nq + n) * 256 + h * 32;
    ctx[base + ml] = (__bf16)(o0[r] * inv);
    ctx[base + 16 + ml] = (__bf16)(o1[r] * inv);
  }
}

// ---------------------------------------------------------------------------
// Kernel 4: output projection + elementwise product with residual input.
// out[s,d] = (ctx(bf16)[s,:] @ W(f32)[:,d] + bias[d]) * mul[s,d]
// The bf16 A slab is staged with CDNA5 async LDS copies (ASYNCcnt-tracked).
// ---------------------------------------------------------------------------
__global__ __launch_bounds__(256) void out_gemm(const __bf16* __restrict__ A,
                                                const float* __restrict__ W,
                                                const float* __restrict__ bias,
                                                const float* __restrict__ mul,
                                                float* __restrict__ out) {
  __shared__ __align__(16) __bf16 As[128][64];
  __shared__ __align__(16) __bf16 Bt[64][64];
  const int tid = threadIdx.x;
  const int row0 = blockIdx.x * 128, col0 = blockIdx.y * 64;
  const int w = tid >> 5, lane = tid & 31;
  const int ml = lane & 15, hi = lane >> 4;
  const int wr = (w & 3) * 32, wc = (w >> 2) * 32;

  v8f acc00 = {0.f,0.f,0.f,0.f,0.f,0.f,0.f,0.f};
  v8f acc01 = acc00, acc10 = acc00, acc11 = acc00;

  for (int ko = 0; ko < 256; ko += 64) {
    __syncthreads();
    // A slab: straight bf16 copy, global -> LDS without touching VGPRs.
    for (int i = tid; i < 1024; i += 256) {
      int r = i >> 3, c = (i & 7) << 3;
      unsigned lds_addr = (unsigned)(unsigned long long)&As[r][c];
      const __bf16* gp = A + (size_t)(row0 + r) * 256 + ko + c;
      asm volatile("global_load_async_to_lds_b128 %0, %1, off"
                   :
                   : "v"(lds_addr), "v"(gp)
                   : "memory");
    }
    // W: transpose + convert (needs VALU, stays on the normal path)
    for (int i = tid; i < 1024; i += 256) {
      int k = i >> 4, j4 = (i & 15) << 2;
      float4 v = *(const float4*)(W + (size_t)(ko + k) * 256 + col0 + j4);
      Bt[j4 + 0][k] = (__bf16)v.x; Bt[j4 + 1][k] = (__bf16)v.y;
      Bt[j4 + 2][k] = (__bf16)v.z; Bt[j4 + 3][k] = (__bf16)v.w;
    }
    asm volatile("s_wait_asynccnt 0x0" ::: "memory");
    __syncthreads();
#pragma unroll
    for (int kk = 0; kk < 64; kk += 32) {
      v16bf a0 = load_frag(&As[wr + ml][kk + hi * 8],      &As[wr + ml][kk + 16 + hi * 8]);
      v16bf a1 = load_frag(&As[wr + 16 + ml][kk + hi * 8], &As[wr + 16 + ml][kk + 16 + hi * 8]);
      v16bf b0 = load_frag(&Bt[wc + ml][kk + hi * 8],      &Bt[wc + ml][kk + 16 + hi * 8]);
      v16bf b1 = load_frag(&Bt[wc + 16 + ml][kk + hi * 8], &Bt[wc + 16 + ml][kk + 16 + hi * 8]);
      acc00 = wmma_bf16(a0, b0, acc00);
      acc01 = wmma_bf16(a0, b1, acc01);
      acc10 = wmma_bf16(a1, b0, acc10);
      acc11 = wmma_bf16(a1, b1, acc11);
    }
  }

  v8f accs[2][2] = {{acc00, acc01}, {acc10, acc11}};
#pragma unroll
  for (int i = 0; i < 2; i++)
#pragma unroll
    for (int j = 0; j < 2; j++)
#pragma unroll
      for (int r = 0; r < 8; r++) {
        int s = row0 + wr + i * 16 + r + 8 * hi;
        int d = col0 + wc + j * 16 + ml;
        size_t idx = (size_t)s * 256 + d;
        out[idx] = (accs[i][j][r] + bias[d]) * mul[idx];
      }
}

// ---------------------------------------------------------------------------
// Host launcher
// ---------------------------------------------------------------------------
extern "C" void kernel_launch(void* const* d_in, const int* in_sizes, int n_in,
                              void* d_out, int out_size, void* d_ws,
                              size_t ws_size, hipStream_t stream) {
  (void)in_sizes; (void)n_in; (void)out_size; (void)ws_size;

  const float* amr_feats = (const float*)d_in[0];
  const int*   amr_mask  = (const int*)d_in[1];
  const float* vis_feats = (const float*)d_in[2];
  const float* W_amr_qk  = (const float*)d_in[3];
  const float* W_amr_v   = (const float*)d_in[4];
  const float* W_vis_qk  = (const float*)d_in[5];
  const float* W_vis_v   = (const float*)d_in[6];
  const float* W_amr_out = (const float*)d_in[7];
  const float* b_amr_out = (const float*)d_in[8];
  const float* W_vis_out = (const float*)d_in[9];
  const float* b_vis_out = (const float*)d_in[10];

  // Workspace layout (bf16 buffers, MB-aligned offsets):
  char* ws = (char*)d_ws;
  __bf16* Qamr  = (__bf16*)(ws + 0);                 //  1 MB: [4,8,512,32]
  __bf16* Kvis  = (__bf16*)(ws + (1ull  << 20));     //  8 MB: [4,8,4096,32]
  __bf16* VamrT = (__bf16*)(ws + (9ull  << 20));     //  1 MB: [4,8,32,512]
  __bf16* VvisT = (__bf16*)(ws + (10ull << 20));     //  8 MB: [4,8,32,4096]
  __bf16* ctxA  = (__bf16*)(ws + (18ull << 20));     //  1 MB: [4,512,256]
  __bf16* ctxV  = (__bf16*)(ws + (19ull << 20));     //  8 MB: [4,4096,256]

  float* outA = (float*)d_out;                       // [4,512,256]
  float* outV = (float*)d_out + 4 * 512 * 256;       // [4,4096,256]

  const dim3 blk(256, 1, 1);
  const float scale = 0.17677669529663687f;  // 1/sqrt(32), folded into Q_amr

  // 1) Projections (fp32 -> bf16, per-head layouts)
  proj_gemm<0, 512> <<<dim3(16, 4),  blk, 0, stream>>>(amr_feats, W_amr_qk, Qamr, scale);
  proj_gemm<0, 4096><<<dim3(128, 4), blk, 0, stream>>>(vis_feats, W_vis_qk, Kvis, 1.0f);
  proj_gemm<1, 512> <<<dim3(16, 4),  blk, 0, stream>>>(amr_feats, W_amr_v, VamrT, 1.0f);
  proj_gemm<1, 4096><<<dim3(128, 4), blk, 0, stream>>>(vis_feats, W_vis_v, VvisT, 1.0f);

  // 2) Flash attention, both directions (scores recomputed, never stored)
  attn_kernel<false><<<dim3(4, 32), blk, 0, stream>>>(
      Qamr, Kvis, VvisT, ctxA, nullptr, 512, 4096);
  attn_kernel<true> <<<dim3(32, 32), blk, 0, stream>>>(
      Kvis, Qamr, VamrT, ctxV, amr_mask, 4096, 512);

  // 3) Output projections + residual elementwise products
  out_gemm<<<dim3(16, 4),  blk, 0, stream>>>(ctxA, W_amr_out, b_amr_out,
                                             amr_feats, outA);
  out_gemm<<<dim3(128, 4), blk, 0, stream>>>(ctxV, W_vis_out, b_vis_out,
                                             vis_feats, outV);
}
